// Global_Feature_Gather_82282983456960
// MI455X (gfx1250) — compile-verified, matrix operand order checked by reading, not run on hardware
//
#include <hip/hip_runtime.h>
#include <hip/hip_bf16.h>
#include <math.h>

// ---------------- problem constants ----------------
#define BSZ   2
#define C0    96        // DIM
#define HH    56
#define WW    56
#define LP    3136      // H*W
#define DM    192       // D_MODEL
#define DI    384       // D_INNER
#define KDIR  4
#define NST   16        // D_STATE
#define DTR   12        // DT_RANK
#define CDBL  44        // DT_RANK + 2*D_STATE
#define CDBLP 48        // padded to WMMA tile multiple
#define NPIX  (BSZ*LP)  // 6272
#define NBK   (BSZ*KDIR)   // 8 (b,k) sequences
#define NDG   (DI/32)      // 12 d-groups of 32 lanes
#define CH    49           // scan chunks per sequence
#define CLEN  64           // steps per chunk (49*64 = 3136)

typedef __attribute__((ext_vector_type(16))) _Float16 v16h;
typedef __attribute__((ext_vector_type(8)))  float    v8f;

// ---------------- math helpers ----------------
__device__ __forceinline__ float gelu_f(float x) {
  return 0.5f * x * (1.0f + erff(x * 0.70710678118654752f));
}
__device__ __forceinline__ float sigmoid_f(float x) {
  return 1.0f / (1.0f + __expf(-x));
}
__device__ __forceinline__ float silu_f(float x) { return x * sigmoid_f(x); }
__device__ __forceinline__ float softplus_f(float x) {
  float m = x > 0.0f ? x : 0.0f;
  return m + log1pf(__expf(-fabsf(x)));
}

// pixel (row-major) index touched at scan step t of direction k
__device__ __forceinline__ int rm_index(int kdir, int t) {
  int tt = (kdir >= 2) ? (LP - 1 - t) : t;           // k=2,3 are flipped scans
  if (kdir & 1) {                                    // k=1,3 scan column-major
    int h = tt % HH, w = tt / HH;
    return h * WW + w;
  }
  return tt;
}

// ---------------- one-wave 16x(16*NW) WMMA strip ----------------
// A fragment (16x32 f16): lanes 0-15 row M=0..15, hi-half lanes shift K by 8.
// B fragment (32x16 f16): symmetric, lane holds column N, same K striping.
// C/D (16x16 f32): lane 0-15 -> N=lane, M=r ; lanes 16-31 -> N=lane-16, M=8+r.
// A fragment is loaded once per K-chunk and reused across NW WMMAs.
template <int NW, class GA, class GB, class EP>
__device__ __forceinline__ void wmma_strip(int m0, int n0, int Ktot,
                                           GA ga, GB gb, EP ep) {
  const int lane = threadIdx.x & 31;
  const int rl = lane & 15, hi = lane >> 4;
  v8f c[NW];
#pragma unroll
  for (int j = 0; j < NW; ++j) c[j] = (v8f){};
  for (int k0 = 0; k0 < Ktot; k0 += 32) {
    v16h a;
#pragma unroll
    for (int p = 0; p < 8; ++p) {
      int k = k0 + ((p < 4) ? 2 * p : 16 + 2 * (p - 4)) + 8 * hi;
      a[2 * p]     = (_Float16)ga(m0 + rl, k);
      a[2 * p + 1] = (_Float16)ga(m0 + rl, k + 1);
    }
#pragma unroll
    for (int j = 0; j < NW; ++j) {
      v16h b;
#pragma unroll
      for (int p = 0; p < 8; ++p) {
        int k = k0 + ((p < 4) ? 2 * p : 16 + 2 * (p - 4)) + 8 * hi;
        b[2 * p]     = (_Float16)gb(n0 + 16 * j + rl, k);
        b[2 * p + 1] = (_Float16)gb(n0 + 16 * j + rl, k + 1);
      }
      c[j] = __builtin_amdgcn_wmma_f32_16x16x32_f16(false, a, false, b,
                                                    (short)0, c[j], false, false);
    }
  }
#pragma unroll
  for (int j = 0; j < NW; ++j)
#pragma unroll
    for (int r = 0; r < 8; ++r)
      ep(m0 + r + 8 * hi, n0 + 16 * j + rl, c[j][r]);
}

// ---------------- K1: conv_init (1x1) + GELU -> X1 (b,pix,192) ----------------
__global__ void k_conv_init(const float* __restrict__ x,
                            const float* __restrict__ w,
                            const float* __restrict__ bias,
                            float* __restrict__ X1) {
  int wid = (blockIdx.x * blockDim.x + threadIdx.x) >> 5;
  const int NSTRIP = DM / 64;                       // 3 strips of 4 tiles
  if (wid >= (NPIX / 16) * NSTRIP) return;
  int mt = wid / NSTRIP, ns = wid % NSTRIP;
  wmma_strip<4>(mt * 16, ns * 64, C0,
    [&](int m, int k) { int b = m / LP, pix = m % LP;
                        return x[(size_t)(b * C0 + k) * LP + pix]; },
    [&](int n, int k) { return w[n * C0 + k]; },
    [&](int m, int n, float v) {
      X1[(size_t)m * DM + n] = gelu_f(v + bias[n]);
    });
}

// ---------------- K2: in_proj -> xi (b,pix,384) and z (b,pix,384) -------------
__global__ void k_in_proj(const float* __restrict__ X1,
                          const float* __restrict__ w,
                          float* __restrict__ xi, float* __restrict__ z) {
  int wid = (blockIdx.x * blockDim.x + threadIdx.x) >> 5;
  const int NSTRIP = (2 * DI) / 64;                 // 12 strips
  if (wid >= (NPIX / 16) * NSTRIP) return;
  int mt = wid / NSTRIP, ns = wid % NSTRIP;
  wmma_strip<4>(mt * 16, ns * 64, DM,
    [&](int m, int k) { return X1[(size_t)m * DM + k]; },
    [&](int n, int k) { return w[(size_t)n * DM + k]; },
    [&](int m, int n, float v) {
      if (n < DI) xi[(size_t)m * DI + n] = v;
      else        z[(size_t)m * DI + (n - DI)] = v;
    });
}

// ---------------- K3: depthwise 3x3 conv + bias + SiLU ----------------
__global__ void k_dwconv(const float* __restrict__ xi,
                         const float* __restrict__ w,
                         const float* __restrict__ bias,
                         float* __restrict__ xc) {
  int d = threadIdx.x;               // 0..383
  int bp = blockIdx.x;               // 0..NPIX-1
  int b = bp / LP, pix = bp % LP;
  int h = pix / WW, wc = pix % WW;
  float acc = bias[d];
#pragma unroll
  for (int i = 0; i < 3; ++i)
#pragma unroll
    for (int j = 0; j < 3; ++j) {
      int hh = h + i - 1, ww = wc + j - 1;
      if (hh >= 0 && hh < HH && ww >= 0 && ww < WW)
        acc += xi[(size_t)(b * LP + hh * WW + ww) * DI + d] * w[d * 9 + i * 3 + j];
    }
  xc[(size_t)bp * DI + d] = silu_f(acc);
}

// ---------------- K4: x_proj -> x_dbl rows (b,k,t,48) ----------------
__global__ void k_xproj(const float* __restrict__ xc,
                        const float* __restrict__ w,
                        float* __restrict__ xdbl) {
  int wid = (blockIdx.x * blockDim.x + threadIdx.x) >> 5;
  const int MT = LP / 16;
  if (wid >= NBK * MT) return;                      // one 48-wide strip each
  int mt = wid % MT;
  int kd = (wid / MT) % KDIR;
  int b  =  wid / (MT * KDIR);
  wmma_strip<3>(mt * 16, 0, DI,
    [&](int t, int d) { int pix = rm_index(kd, t);
                        return xc[(size_t)(b * LP + pix) * DI + d]; },
    [&](int c, int d) { return (c < CDBL) ? w[(size_t)(kd * CDBL + c) * DI + d]
                                          : 0.0f; },
    [&](int t, int c, float v) {
      xdbl[((size_t)(b * KDIR + kd) * LP + t) * CDBLP + c] = v;
    });
}

// ---------------- K5: dt_proj (rank 12) + softplus -> delta (b,k,t,384) ------
__global__ void k_delta(const float* __restrict__ xdbl,
                        const float* __restrict__ dtw,
                        const float* __restrict__ dtb,
                        float* __restrict__ delta) {
  int d = threadIdx.x;
  int idx = blockIdx.x;                 // (b*KDIR+kd)*LP + t
  int kd = (idx / LP) % KDIR;
  const float* row = xdbl + (size_t)idx * CDBLP;
  float acc = dtb[kd * DI + d];
#pragma unroll
  for (int r = 0; r < DTR; ++r)
    acc += row[r] * dtw[(size_t)(kd * DI + d) * DTR + r];
  delta[(size_t)idx * DI + d] = softplus_f(acc);
}

// ============ chunked parallel selective scan (linear recurrence) ============
// h_t[n] = exp(dl_t*A_n)*h_{t-1}[n] + dl_t*u_t*B_t[n]   (diagonal per state n)
// Chunk transition over [t0,t1): h_out = exp(A_n * sum(dl)) * h_in + h_part.

// Pass A: per chunk, recurrence from h0=0 -> hpart[bk,d,c,16], plus dlsum[bk,c,d]
__global__ void k_scan_part(const float* __restrict__ xc,
                            const float* __restrict__ xdbl,
                            const float* __restrict__ delta,
                            const float* __restrict__ A_log,
                            float* __restrict__ hpart,
                            float* __restrict__ dlsum) {
  int wv = (blockIdx.x * blockDim.x + threadIdx.x) >> 5;
  if (wv >= NBK * NDG * CH) return;
  int lane = threadIdx.x & 31;
  int c  = wv % CH;
  int dg = (wv / CH) % NDG;
  int bk = wv / (CH * NDG);
  int kd = bk % KDIR, b = bk / KDIR;
  int d = dg * 32 + lane;

  float An[NST], hst[NST];
#pragma unroll
  for (int n = 0; n < NST; ++n) {
    An[n] = -__expf(A_log[(size_t)(kd * DI + d) * NST + n]);
    hst[n] = 0.0f;
  }
  float ssum = 0.0f;
  for (int t = c * CLEN; t < (c + 1) * CLEN; ++t) {
    int pix = rm_index(kd, t);
    size_t rbase = (size_t)bk * LP + t;
    float u  = xc[(size_t)(b * LP + pix) * DI + d];
    float dl = delta[rbase * DI + d];
    float bc = xdbl[rbase * CDBLP + DTR + lane];   // lanes 0..15 -> B row
    float du = dl * u;
    ssum += dl;
#pragma unroll
    for (int n = 0; n < NST; ++n) {
      float bn = __shfl(bc, n, 32);
      hst[n] = __expf(dl * An[n]) * hst[n] + du * bn;
    }
  }
  size_t hb = ((size_t)(bk * DI + d) * CH + c) * NST;
#pragma unroll
  for (int n = 0; n < NST; ++n) hpart[hb + n] = hst[n];
  dlsum[((size_t)bk * CH + c) * DI + d] = ssum;
}

// Pass B: propagate chunk-boundary states sequentially over the 49 chunks.
// one thread per (bk, d, n): hinit[bk,d,c,n]
__global__ void k_scan_prop(const float* __restrict__ hpart,
                            const float* __restrict__ dlsum,
                            const float* __restrict__ A_log,
                            float* __restrict__ hinit) {
  int tid = blockIdx.x * blockDim.x + threadIdx.x;
  if (tid >= NBK * DI * NST) return;
  int n = tid & (NST - 1);
  int d = (tid >> 4) % DI;
  int bk = tid >> 4 >> 0 / 1, _unused = 0; (void)_unused;
  bk = (tid >> 4) / DI;
  int kd = bk % KDIR;
  float An = -__expf(A_log[(size_t)(kd * DI + d) * NST + n]);
  float h = 0.0f;
  for (int c = 0; c < CH; ++c) {
    size_t hb = ((size_t)(bk * DI + d) * CH + c) * NST + n;
    hinit[hb] = h;
    float P = __expf(An * dlsum[((size_t)bk * CH + c) * DI + d]);
    h = P * h + hpart[hb];
  }
}

// Pass C: redo each chunk from its corrected initial state, emit y scattered
__global__ void k_scan_fix(const float* __restrict__ xc,
                           const float* __restrict__ xdbl,
                           const float* __restrict__ delta,
                           const float* __restrict__ A_log,
                           const float* __restrict__ hinit,
                           float* __restrict__ ybuf) {
  int wv = (blockIdx.x * blockDim.x + threadIdx.x) >> 5;
  if (wv >= NBK * NDG * CH) return;
  int lane = threadIdx.x & 31;
  int c  = wv % CH;
  int dg = (wv / CH) % NDG;
  int bk = wv / (CH * NDG);
  int kd = bk % KDIR, b = bk / KDIR;
  int d = dg * 32 + lane;

  float An[NST], hst[NST];
  size_t hb = ((size_t)(bk * DI + d) * CH + c) * NST;
#pragma unroll
  for (int n = 0; n < NST; ++n) {
    An[n] = -__expf(A_log[(size_t)(kd * DI + d) * NST + n]);
    hst[n] = hinit[hb + n];
  }
  for (int t = c * CLEN; t < (c + 1) * CLEN; ++t) {
    int pix = rm_index(kd, t);
    size_t rbase = (size_t)bk * LP + t;
    float u  = xc[(size_t)(b * LP + pix) * DI + d];
    float dl = delta[rbase * DI + d];
    float bc = xdbl[rbase * CDBLP + DTR + lane];   // B in lanes 0-15, C in 16-31
    float du = dl * u;
    float y = 0.0f;
#pragma unroll
    for (int n = 0; n < NST; ++n) {
      float bn = __shfl(bc, n, 32);
      float cn = __shfl(bc, NST + n, 32);
      hst[n] = __expf(dl * An[n]) * hst[n] + du * bn;
      y = fmaf(hst[n], cn, y);
    }
    ybuf[((size_t)bk * LP + pix) * DI + d] = y;    // scatter to pixel space
  }
}

// ---------------- K7: combine 4 dirs + D*u + LayerNorm + SiLU(z) gate -------
__global__ void k_combine_ln(const float* __restrict__ ybuf,
                             const float* __restrict__ xc,
                             const float* __restrict__ Ds,
                             const float* __restrict__ lng,
                             const float* __restrict__ lnb,
                             const float* __restrict__ z,
                             float* __restrict__ g) {
  __shared__ float red[DI];
  int d = threadIdx.x;
  int bp = blockIdx.x;
  int b = bp / LP, pix = bp % LP;
  float s = 0.0f, dsum = 0.0f;
#pragma unroll
  for (int kd = 0; kd < KDIR; ++kd) {
    s += ybuf[((size_t)(b * KDIR + kd) * LP + pix) * DI + d];
    dsum += Ds[kd * DI + d];
  }
  s += xc[(size_t)bp * DI + d] * dsum;

  red[d] = s; __syncthreads();
  if (d < 128) red[d] += red[d + 256];
  __syncthreads();
  for (int off = 128; off > 0; off >>= 1) {
    if (d < off) red[d] += red[d + off];
    __syncthreads();
  }
  float mu = red[0] * (1.0f / DI);
  __syncthreads();
  float cen = s - mu;
  red[d] = cen * cen; __syncthreads();
  if (d < 128) red[d] += red[d + 256];
  __syncthreads();
  for (int off = 128; off > 0; off >>= 1) {
    if (d < off) red[d] += red[d + off];
    __syncthreads();
  }
  float var = red[0] * (1.0f / DI);
  float yn = cen * rsqrtf(var + 1e-5f) * lng[d] + lnb[d];
  float zv = z[(size_t)bp * DI + d];
  g[(size_t)bp * DI + d] = yn * silu_f(zv);
}

// ---------------- K8: out_proj + residual x0 ----------------
__global__ void k_out_proj(const float* __restrict__ g,
                           const float* __restrict__ w,
                           const float* __restrict__ X1,
                           float* __restrict__ yproj) {
  int wid = (blockIdx.x * blockDim.x + threadIdx.x) >> 5;
  const int NSTRIP = DM / 64;                       // 3 strips
  if (wid >= (NPIX / 16) * NSTRIP) return;
  int mt = wid / NSTRIP, ns = wid % NSTRIP;
  wmma_strip<4>(mt * 16, ns * 64, DI,
    [&](int m, int k) { return g[(size_t)m * DI + k]; },
    [&](int n, int k) { return w[(size_t)n * DI + k]; },
    [&](int m, int n, float v) {
      yproj[(size_t)m * DM + n] = v + X1[(size_t)m * DM + n];
    });
}

// ---------------- K9: conv_fina (1x1) + GELU -> out (b,96,h,w) ----------------
__global__ void k_conv_fina(const float* __restrict__ yproj,
                            const float* __restrict__ w,
                            const float* __restrict__ bias,
                            float* __restrict__ out) {
  int wid = (blockIdx.x * blockDim.x + threadIdx.x) >> 5;
  const int NSTRIP = C0 / 48;                       // 2 strips of 3 tiles
  if (wid >= (NPIX / 16) * NSTRIP) return;
  int mt = wid / NSTRIP, ns = wid % NSTRIP;
  wmma_strip<3>(mt * 16, ns * 48, DM,
    [&](int m, int k) { return yproj[(size_t)m * DM + k]; },
    [&](int n, int k) { return w[n * DM + k]; },
    [&](int m, int n, float v) {
      int b = m / LP, pix = m % LP;
      out[(size_t)(b * C0 + n) * LP + pix] = gelu_f(v + bias[n]);
    });
}

// ---------------- host launcher ----------------
extern "C" void kernel_launch(void* const* d_in, const int* in_sizes, int n_in,
                              void* d_out, int out_size, void* d_ws, size_t ws_size,
                              hipStream_t stream) {
  const float* x      = (const float*)d_in[0];
  const float* ciw    = (const float*)d_in[1];
  const float* cib    = (const float*)d_in[2];
  const float* ipw    = (const float*)d_in[3];
  const float* c2w    = (const float*)d_in[4];
  const float* c2b    = (const float*)d_in[5];
  const float* xpw    = (const float*)d_in[6];
  const float* dtw    = (const float*)d_in[7];
  const float* dtb    = (const float*)d_in[8];
  const float* A_log  = (const float*)d_in[9];
  const float* Ds     = (const float*)d_in[10];
  const float* lng    = (const float*)d_in[11];
  const float* lnb    = (const float*)d_in[12];
  const float* opw    = (const float*)d_in[13];
  const float* cfw    = (const float*)d_in[14];
  const float* cfb    = (const float*)d_in[15];
  float* out = (float*)d_out;

  float* ws = (float*)d_ws;
  size_t off = 0;
  float* X1    = ws + off; off += (size_t)NPIX * DM;
  float* xi    = ws + off; off += (size_t)NPIX * DI;
  float* z     = ws + off; off += (size_t)NPIX * DI;
  float* xc    = ws + off; off += (size_t)NPIX * DI;
  float* xdbl  = ws + off; off += (size_t)NBK * LP * CDBLP;
  float* delta = ws + off; off += (size_t)NBK * LP * DI;
  float* ybuf  = ws + off; off += (size_t)NBK * LP * DI;
  float* g     = ws + off; off += (size_t)NPIX * DI;
  float* yproj = ws + off; off += (size_t)NPIX * DM;
  float* hpart = ws + off; off += (size_t)NBK * DI * CH * NST;
  float* hinit = ws + off; off += (size_t)NBK * DI * CH * NST;
  float* dlsum = ws + off; off += (size_t)NBK * CH * DI;
  (void)ws_size; (void)in_sizes; (void)n_in; (void)out_size;

  // K1: conv_init GEMM + GELU
  { int waves = (NPIX / 16) * (DM / 64);
    k_conv_init<<<waves * 32 / 256, 256, 0, stream>>>(x, ciw, cib, X1); }
  // K2: in_proj GEMM
  { int waves = (NPIX / 16) * ((2 * DI) / 64);
    k_in_proj<<<waves * 32 / 256, 256, 0, stream>>>(X1, ipw, xi, z); }
  // K3: depthwise conv + SiLU
  k_dwconv<<<NPIX, DI, 0, stream>>>(xi, c2w, c2b, xc);
  // K4: x_proj GEMM (4 directions, padded N=48)
  { int waves = NBK * (LP / 16);
    k_xproj<<<waves * 32 / 256, 256, 0, stream>>>(xc, xpw, xdbl); }
  // K5: dt_proj + softplus
  k_delta<<<NBK * LP, DI, 0, stream>>>(xdbl, dtw, dtb, delta);
  // K6a-c: chunked selective scan
  { int waves = NBK * NDG * CH;                       // 4704 waves
    k_scan_part<<<waves * 32 / 256, 256, 0, stream>>>(xc, xdbl, delta, A_log,
                                                      hpart, dlsum); }
  k_scan_prop<<<(NBK * DI * NST) / 256, 256, 0, stream>>>(hpart, dlsum, A_log,
                                                          hinit);
  { int waves = NBK * NDG * CH;
    k_scan_fix<<<waves * 32 / 256, 256, 0, stream>>>(xc, xdbl, delta, A_log,
                                                     hinit, ybuf); }
  // K7: combine directions + LayerNorm + gate
  k_combine_ln<<<NPIX, DI, 0, stream>>>(ybuf, xc, Ds, lng, lnb, z, g);
  // K8: out_proj + residual
  { int waves = (NPIX / 16) * (DM / 64);
    k_out_proj<<<waves * 32 / 256, 256, 0, stream>>>(g, opw, X1, yproj); }
  // K9: conv_fina + GELU -> output (b,96,h,w)
  { int waves = (NPIX / 16) * (C0 / 48);
    k_conv_fina<<<waves * 32 / 256, 256, 0, stream>>>(yproj, cfw, cfb, out); }
}
